// GroupedQueryAttention_54984171323577
// MI455X (gfx1250) — matrix operand order repaired and checked by reading, not compiled, not run on hardware
//
#include <hip/hip_runtime.h>

// GQA attention for MI455X (gfx1250), bf16 WMMA pipeline with f32 accumulate.
// Shapes: b=2, s=2048, D=2048, 8 Q heads / 4 KV heads, head_dim=256.

#define D_IN    2048
#define SEQ     2048
#define BATCH   2
#define NH      8
#define NKV     4
#define HD      256
#define SCALE_Q 0.0625f   // 256^-0.5

typedef __attribute__((ext_vector_type(16))) __bf16   bf16x16;
typedef __attribute__((ext_vector_type(8)))  float    f32x8;
typedef __attribute__((ext_vector_type(4)))  unsigned uv4;
typedef int i32x4 __attribute__((vector_size(4 * sizeof(int))));

#define AS1 __attribute__((address_space(1)))
#define AS3 __attribute__((address_space(3)))

// gfx1250 async global->LDS DMA path (ASYNCcnt), if the toolchain exposes it.
#if defined(__gfx1250__) && \
    __has_builtin(__builtin_amdgcn_global_load_async_to_lds_b128) && \
    __has_builtin(__builtin_amdgcn_s_wait_asynccnt)
#define HAVE_ASYNC_LDS 1
#else
#define HAVE_ASYNC_LDS 0
#endif

#if HAVE_ASYNC_LDS
// src: global 16B granule, dst: LDS 16B granule.
#define ASYNC_CP16(srcp, dstp) \
  __builtin_amdgcn_global_load_async_to_lds_b128( \
      (AS1 i32x4*)(srcp), (AS3 i32x4*)(dstp), 0, 0)
#endif

union Frag {
  bf16x16 v;
  uv4 u[2];
};

__device__ __forceinline__ f32x8 wmma_bf16(const Frag& a, const Frag& b, f32x8 c) {
  // (neg_a, A, neg_b, B, c_mod, C, reuse_a, reuse_b)
  return __builtin_amdgcn_wmma_f32_16x16x32_bf16(false, a.v, false, b.v,
                                                 (short)0, c, false, false);
}

// ---------------------------------------------------------------------------
// fp32 -> bf16 streaming convert
// ---------------------------------------------------------------------------
__global__ __launch_bounds__(256)
void f32_to_bf16_kernel(__bf16* __restrict__ dst, const float* __restrict__ src,
                        long n) {
  long i = (long)blockIdx.x * blockDim.x + threadIdx.x;
  long stride = (long)gridDim.x * blockDim.x;
  for (; i < n; i += stride) dst[i] = (__bf16)src[i];
}

// ---------------------------------------------------------------------------
// C[M,N] (f32) = A[M,K] (bf16, row-major) * B[N,K]^T (bf16, row-major weights)
// Block tile 128x128, k-step 32, 8 waves (4 along M x 2 along N), wave 32x64.
// ---------------------------------------------------------------------------
__global__ __launch_bounds__(256)
void gemm_bf16_wmma_kernel(float* __restrict__ C, const __bf16* __restrict__ A,
                           const __bf16* __restrict__ Bw, int M, int N, int K) {
  __shared__ __attribute__((aligned(16))) __bf16 As[128][32];
  __shared__ __attribute__((aligned(16))) __bf16 Bs[128][32];

  const int bm = blockIdx.y * 128;
  const int bn = blockIdx.x * 128;
  const int tid  = threadIdx.x;
  const int wave = tid >> 5;
  const int lane = tid & 31;
  const int wm = (wave >> 1) * 32;
  const int wn = (wave & 1) * 64;
  const int l15  = lane & 15;
  const int akb  = (lane >> 4) * 8;    // A-frag K chunk offset
  const int bkb  = (lane >> 4) * 16;   // unused; B-frag loads use akb too
  const int rofs = (lane >> 4) * 8;    // C/D row offset

  f32x8 acc[2][4] = {};

  const int lr = tid >> 1;             // 0..127 tile row per loader thread
  const int lc = (tid & 1) * 16;       // 0 or 16

  for (int k0 = 0; k0 < K; k0 += 32) {
#if HAVE_ASYNC_LDS
    ASYNC_CP16(&A [(size_t)(bm + lr) * K + k0 + lc],     &As[lr][lc]);
    ASYNC_CP16(&A [(size_t)(bm + lr) * K + k0 + lc + 8], &As[lr][lc + 8]);
    ASYNC_CP16(&Bw[(size_t)(bn + lr) * K + k0 + lc],     &Bs[lr][lc]);
    ASYNC_CP16(&Bw[(size_t)(bn + lr) * K + k0 + lc + 8], &Bs[lr][lc + 8]);
    __builtin_amdgcn_s_wait_asynccnt(0);
#else
    *(uv4*)&As[lr][lc]     = *(const uv4*)&A [(size_t)(bm + lr) * K + k0 + lc];
    *(uv4*)&As[lr][lc + 8] = *(const uv4*)&A [(size_t)(bm + lr) * K + k0 + lc + 8];
    *(uv4*)&Bs[lr][lc]     = *(const uv4*)&Bw[(size_t)(bn + lr) * K + k0 + lc];
    *(uv4*)&Bs[lr][lc + 8] = *(const uv4*)&Bw[(size_t)(bn + lr) * K + k0 + lc + 8];
#endif
    __syncthreads();
#pragma unroll
    for (int i = 0; i < 2; ++i) {
      Frag af;
      af.u[0] = *(const uv4*)&As[wm + 16 * i + l15][akb];
      af.u[1] = *(const uv4*)&As[wm + 16 * i + l15][akb + 16];
#pragma unroll
      for (int j = 0; j < 4; ++j) {
        Frag bf;
        bf.u[0] = *(const uv4*)&Bs[wn + 16 * j + l15][akb];
        bf.u[1] = *(const uv4*)&Bs[wn + 16 * j + l15][akb + 16];
        acc[i][j] = wmma_bf16(af, bf, acc[i][j]);
      }
    }
    __syncthreads();
  }

#pragma unroll
  for (int i = 0; i < 2; ++i)
#pragma unroll
    for (int j = 0; j < 4; ++j) {
      const int col = bn + wn + 16 * j + l15;
#pragma unroll
      for (int r = 0; r < 8; ++r) {
        const int row = bm + wm + 16 * i + rofs + r;
        C[(size_t)row * N + col] = acc[i][j][r];
      }
    }
}

// ---------------------------------------------------------------------------
// QKV epilogue: RMSnorm(+gamma) and RoPE for Q/K (Q also scaled), V passthru.
// grid.x = token (b*2048+s), grid.y = slot (0-7 Q heads, 8-11 K, 12-15 V).
// Q -> Qb[b][h][s][d] bf16, K -> Kb[b][kv][s][d] bf16, V -> Vt[b][kv][d][s] bf16.
// ---------------------------------------------------------------------------
__global__ __launch_bounds__(256)
void qkv_post_kernel(const float* __restrict__ qkvf, const float* __restrict__ cosb,
                     const float* __restrict__ sinb, const float* __restrict__ qg,
                     const float* __restrict__ kg, __bf16* __restrict__ Qb,
                     __bf16* __restrict__ Kb, __bf16* __restrict__ Vt) {
  __shared__ float red[256];
  const int t    = blockIdx.x;
  const int slot = blockIdx.y;
  const int d    = threadIdx.x;
  const int bi = t >> 11, sp = t & 2047;

  const float* row;
  if (slot < 8)       row = qkvf + (size_t)t * 4096 + slot * 256;
  else if (slot < 12) row = qkvf + (size_t)t * 4096 + 2048 + (slot - 8) * 256;
  else                row = qkvf + (size_t)t * 4096 + 3072 + (slot - 12) * 256;

  const float x = row[d];
  if (slot >= 12) {  // V: no norm/rope; store transposed [d][s]
    Vt[((size_t)(bi * NKV + (slot - 12)) * HD + d) * SEQ + sp] = (__bf16)x;
    return;
  }

  red[d] = x * x;
  __syncthreads();
  for (int off = 128; off > 0; off >>= 1) {
    if (d < off) red[d] += red[d + off];
    __syncthreads();
  }
  const float inv = rsqrtf(red[0] * (1.0f / 256.0f) + 1e-6f);

  const int   p  = (d < 128) ? d + 128 : d - 128;
  const float xp = row[p];
  const float* g = (slot < 8) ? qg : kg;
  const float xn  = x  * inv * (1.0f + g[d]);
  const float xpn = xp * inv * (1.0f + g[p]);
  const float c = cosb[(size_t)sp * HD + d];
  const float s = sinb[(size_t)sp * HD + d];
  const float rot = (d < 128) ? -xpn : xpn;
  const float out = xn * c + rot * s;

  if (slot < 8)
    Qb[((size_t)(bi * NH + slot) * SEQ + sp) * HD + d] = (__bf16)(out * SCALE_Q);
  else
    Kb[((size_t)(bi * NKV + (slot - 8)) * SEQ + sp) * HD + d] = (__bf16)out;
}

// ---------------------------------------------------------------------------
// Flash-style causal attention.
// grid.x = b*NH+h (16), grid.y = query block of 64 (32). 128 threads = 4 waves,
// each wave owns 16 query rows. 32-key steps staged in LDS; online softmax.
// ctx -> ctxb[token][h*256+d] bf16 (feeds output GEMM).
// ---------------------------------------------------------------------------
__global__ __launch_bounds__(128)
void attn_kernel(const __bf16* __restrict__ Qb, const __bf16* __restrict__ Kb,
                 const __bf16* __restrict__ Vt, __bf16* __restrict__ ctx) {
  __shared__ __attribute__((aligned(16))) __bf16 Ks[32][256];   // [key][dim]
  __shared__ __attribute__((aligned(16))) __bf16 Vs[256][32];   // [dim][key]
  __shared__ __attribute__((aligned(16))) __bf16 Ps[4][16][32]; // per-wave P^T stage

  const int bh   = blockIdx.x;           // b*NH + h
  const int qblk = blockIdx.y;
  const int b = bh >> 3, h = bh & 7, kv = h >> 1;
  const int tid = threadIdx.x, wave = tid >> 5, lane = tid & 31;
  const int l15  = lane & 15;
  const int akb  = (lane >> 4) * 8;
  const int bkb  = (lane >> 4) * 16;
  const int rofs = (lane >> 4) * 8;
  const int q0   = qblk * 64 + wave * 16;

  const __bf16* Qp = Qb + ((size_t)bh * SEQ + q0) * HD;
  const __bf16* Kp = Kb + (size_t)(b * NKV + kv) * SEQ * HD;
  const __bf16* Vp = Vt + (size_t)(b * NKV + kv) * HD * SEQ;

  // Preload the wave's 16x256 Q block as 8 A-fragments (k-chunks of 32).
  Frag qf[8];
#pragma unroll
  for (int f = 0; f < 8; ++f) {
    const __bf16* p = Qp + (size_t)l15 * HD + f * 32 + akb;
    qf[f].u[0] = *(const uv4*)p;
    qf[f].u[1] = *(const uv4*)(p + 16);
  }

  f32x8 acc[16] = {};          // 16 x (16x16) ctx tiles covering head_dim=256
  float mrow[8], lrow[8];
#pragma unroll
  for (int r = 0; r < 8; ++r) { mrow[r] = -1e30f; lrow[r] = 0.0f; }

  const int kend = qblk * 64 + 64;
  for (int kb0 = 0; kb0 < kend; kb0 += 32) {
    const __bf16* Kt = Kp + (size_t)kb0 * HD;
#if HAVE_ASYNC_LDS
#pragma unroll
    for (int u = 0; u < 8; ++u) {
      const int e = (tid + 128 * u) * 8;   // bf16 element index, 16B granules
      ASYNC_CP16(Kt + e, &Ks[0][0] + e);
    }
#pragma unroll
    for (int u = 0; u < 8; ++u) {
      const int idx = tid + 128 * u;
      const int dd = idx >> 2, cc = idx & 3;
      ASYNC_CP16(Vp + (size_t)dd * SEQ + kb0 + cc * 8, &Vs[dd][cc * 8]);
    }
#else
    const uv4* Ktu = (const uv4*)Kt;
    uv4* Ksu = (uv4*)&Ks[0][0];
#pragma unroll
    for (int u = 0; u < 8; ++u) Ksu[tid + 128 * u] = Ktu[tid + 128 * u];
#pragma unroll
    for (int u = 0; u < 8; ++u) {
      const int idx = tid + 128 * u;
      const int dd = idx >> 2, cc = idx & 3;
      *(uv4*)&Vs[dd][cc * 8] = *(const uv4*)(Vp + (size_t)dd * SEQ + kb0 + cc * 8);
    }
#endif
    if (kb0 + 32 < kend)  // gfx1250 global_prefetch_b8 for next K tile
      __builtin_prefetch(Kt + 32 * HD + tid * 64, 0, 1);
#if HAVE_ASYNC_LDS
    __builtin_amdgcn_s_wait_asynccnt(0);
#endif
    __syncthreads();

    // Scores: two 16x16 tiles over 32 keys, K-dim reduction 256 = 8 WMMAs each.
    f32x8 sc[2] = {};
#pragma unroll
    for (int nt = 0; nt < 2; ++nt) {
      const int kc = nt * 16 + l15;
#pragma unroll
      for (int f = 0; f < 8; ++f) {
        Frag kf;
        kf.u[0] = *(const uv4*)&Ks[kc][f * 32 + akb];
        kf.u[1] = *(const uv4*)&Ks[kc][f * 32 + akb + 16];
        sc[nt] = wmma_bf16(qf[f], kf, sc[nt]);
      }
    }

    // Causal mask + online softmax (row lives across 16 lanes of a half-wave).
    float alpha[8];
#pragma unroll
    for (int r = 0; r < 8; ++r) {
      const int qi  = q0 + rofs + r;
      const int ki0 = kb0 + l15;
      if (ki0 > qi)      sc[0][r] = -1e30f;
      if (ki0 + 16 > qi) sc[1][r] = -1e30f;
      float m = fmaxf(sc[0][r], sc[1][r]);
      for (int off = 8; off > 0; off >>= 1) m = fmaxf(m, __shfl_xor(m, off, 16));
      const float mn = fmaxf(mrow[r], m);
      const float a  = __expf(mrow[r] - mn);
      const float p0 = (sc[0][r] < -1e29f) ? 0.0f : __expf(sc[0][r] - mn);
      const float p1 = (sc[1][r] < -1e29f) ? 0.0f : __expf(sc[1][r] - mn);
      float ls = p0 + p1;
      for (int off = 8; off > 0; off >>= 1) ls += __shfl_xor(ls, off, 16);
      lrow[r]  = lrow[r] * a + ls;
      mrow[r]  = mn;
      alpha[r] = a;
      sc[0][r] = p0;
      sc[1][r] = p1;
    }
#pragma unroll
    for (int t2 = 0; t2 < 16; ++t2)
#pragma unroll
      for (int r = 0; r < 8; ++r) acc[t2][r] *= alpha[r];

    // Transpose P (C-layout) into A-fragment layout through per-wave LDS.
#pragma unroll
    for (int nt = 0; nt < 2; ++nt)
#pragma unroll
      for (int r = 0; r < 8; ++r)
        Ps[wave][rofs + r][nt * 16 + l15] = (__bf16)sc[nt][r];

    Frag pf;
    pf.u[0] = *(const uv4*)&Ps[wave][l15][akb];
    pf.u[1] = *(const uv4*)&Ps[wave][l15][akb + 16];

    // PV: 16 ctx tiles, one WMMA each over the 32-key reduction.
#pragma unroll
    for (int t2 = 0; t2 < 16; ++t2) {
      Frag vf;
      vf.u[0] = *(const uv4*)&Vs[t2 * 16 + l15][akb];
      vf.u[1] = *(const uv4*)&Vs[t2 * 16 + l15][akb + 16];
      acc[t2] = wmma_bf16(pf, vf, acc[t2]);
    }
    __syncthreads();
  }

  // Normalize by l and store ctx in [token][h*256+d] for the output GEMM.
#pragma unroll
  for (int t2 = 0; t2 < 16; ++t2) {
    const int dc = t2 * 16 + l15;
#pragma unroll
    for (int r = 0; r < 8; ++r) {
      const int qi = q0 + rofs + r;
      const float val = acc[t2][r] / fmaxf(lrow[r], 1e-20f);
      ctx[((size_t)(b * SEQ + qi)) * (NH * HD) + h * HD + dc] = (__bf16)val;
    }
  }
}

// ---------------------------------------------------------------------------
// Host launcher
// ---------------------------------------------------------------------------
extern "C" void kernel_launch(void* const* d_in, const int* in_sizes, int n_in,
                              void* d_out, int out_size, void* d_ws, size_t ws_size,
                              hipStream_t stream) {
  (void)in_sizes; (void)n_in; (void)out_size; (void)ws_size;
  const float* x    = (const float*)d_in[0];
  // d_in[1] = mask (bool) — causal mask is computed analytically, unused.
  const float* cosb = (const float*)d_in[2];
  const float* sinb = (const float*)d_in[3];
  const float* wq   = (const float*)d_in[4];
  const float* wk   = (const float*)d_in[5];
  const float* wv   = (const float*)d_in[6];
  const float* wo   = (const float*)d_in[7];
  const float* qg   = (const float*)d_in[8];
  const float* kg   = (const float*)d_in[9];

  char* ws = (char*)d_ws;
  size_t off = 0;
  auto alloc = [&](size_t bytes) -> char* {
    char* p = ws + off;
    off += (bytes + 255) & ~(size_t)255;
    return p;
  };
  const size_t TOK = (size_t)BATCH * SEQ;                       // 4096 tokens
  __bf16* xb   = (__bf16*)alloc(TOK * D_IN * 2);                // 16 MB
  __bf16* wb   = (__bf16*)alloc((size_t)4096 * D_IN * 2);       // 16 MB (wq|wk|wv)
  __bf16* wob  = (__bf16*)alloc((size_t)D_IN * D_IN * 2);       //  8 MB
  float*  qkvf = (float* )alloc(TOK * 4096 * 4);                // 64 MB
  __bf16* Qb   = (__bf16*)alloc((size_t)BATCH * NH  * SEQ * HD * 2);  // 16 MB
  __bf16* Kb   = (__bf16*)alloc((size_t)BATCH * NKV * SEQ * HD * 2);  //  8 MB
  __bf16* Vt   = (__bf16*)alloc((size_t)BATCH * NKV * HD * SEQ * 2);  //  8 MB
  __bf16* ctxb = (__bf16*)alloc(TOK * (size_t)(NH * HD) * 2);         // 16 MB

  // 1) bf16 converts
  f32_to_bf16_kernel<<<2048, 256, 0, stream>>>(xb, x, (long)TOK * D_IN);
  f32_to_bf16_kernel<<<2048, 256, 0, stream>>>(wb, wq, (long)2048 * D_IN);
  f32_to_bf16_kernel<<<1024, 256, 0, stream>>>(wb + (size_t)2048 * D_IN, wk,
                                               (long)1024 * D_IN);
  f32_to_bf16_kernel<<<1024, 256, 0, stream>>>(wb + (size_t)3072 * D_IN, wv,
                                               (long)1024 * D_IN);
  f32_to_bf16_kernel<<<2048, 256, 0, stream>>>(wob, wo, (long)D_IN * D_IN);

  // 2) fused QKV projection: [4096 x 4096] = xb[4096 x 2048] * wb^T
  gemm_bf16_wmma_kernel<<<dim3(4096 / 128, 4096 / 128), 256, 0, stream>>>(
      qkvf, xb, wb, 4096, 4096, D_IN);

  // 3) RMSnorm + RoPE epilogue, V transpose
  qkv_post_kernel<<<dim3(4096, 16), 256, 0, stream>>>(qkvf, cosb, sinb, qg, kg,
                                                      Qb, Kb, Vt);

  // 4) causal flash attention
  attn_kernel<<<dim3(BATCH * NH, SEQ / 64), 128, 0, stream>>>(Qb, Kb, Vt, ctxb);

  // 5) output projection: d_out[4096 x 2048] = ctxb[4096 x 2048] * wo^T
  gemm_bf16_wmma_kernel<<<dim3(D_IN / 128, 4096 / 128), 256, 0, stream>>>(
      (float*)d_out, ctxb, wob, 4096, D_IN, D_IN);
}
